// VectorizedSplatAttention_43224550867597
// MI455X (gfx1250) — matrix operand order repaired and checked by reading, not compiled, not run on hardware
//
#include <hip/hip_runtime.h>

// ---------------------------------------------------------------------------
// VectorizedSplatAttention for MI455X (gfx1250), wave32 + WMMA f16->f32.
// B=2, S=2048, D=1024, H=16, K=16 splats, hd=64.
// - Static-shift softmax (scores bounded in [0,16]) -> no online max.
// - GEMMs: double-buffered LDS fed by CDNA5 async global->LDS copies
//   (GLOBAL_LOAD_ASYNC_TO_LDS_B128, ASYNCcnt) instead of register staging.
// ---------------------------------------------------------------------------

typedef __attribute__((ext_vector_type(16))) _Float16 v16h;
typedef __attribute__((ext_vector_type(8)))  float    v8f;
typedef _Float16 half_t;

static constexpr int Bc  = 2;
static constexpr int Sc  = 2048;
static constexpr int Dc  = 1024;
static constexpr int Hc  = 16;
static constexpr int Kc  = 16;
static constexpr int HDc = 64;
static constexpr int KP  = 32;   // splat K padded to one WMMA K-step

// --- CDNA5 async global->LDS copy helpers ----------------------------------
// LDS byte address = low 32 bits of the generic pointer (ISA: aperture rule).
__device__ __forceinline__ void async_copy_b128(void* lds_ptr, const void* gptr) {
  unsigned lds_addr = (unsigned)(uintptr_t)lds_ptr;
  unsigned long long ga = (unsigned long long)(uintptr_t)gptr;
  asm volatile("global_load_async_to_lds_b128 %0, %1, off"
               :: "v"(lds_addr), "v"(ga) : "memory");
}

__device__ __forceinline__ void wait_async_zero() {
#if __has_builtin(__builtin_amdgcn_s_wait_asynccnt)
  __builtin_amdgcn_s_wait_asynccnt(0);
#else
  asm volatile("s_wait_asynccnt 0x0" ::: "memory");
#endif
}

// ---------------------------------------------------------------------------
// f32 -> f16 cast, 4 elements / thread
// ---------------------------------------------------------------------------
__global__ __launch_bounds__(256) void cast_f32_to_f16(const float* __restrict__ in,
                                                       half_t* __restrict__ out,
                                                       int n4) {
  int i = blockIdx.x * blockDim.x + threadIdx.x;
  if (i >= n4) return;
  float4 v = reinterpret_cast<const float4*>(in)[i];
  half_t* o = out + (size_t)i * 4;
  o[0] = (half_t)v.x; o[1] = (half_t)v.y; o[2] = (half_t)v.z; o[3] = (half_t)v.w;
}

// ---------------------------------------------------------------------------
// Trajectory pass 1: per (b,j) diff magnitude -> tanh weight + inverse norm.
// ---------------------------------------------------------------------------
__global__ __launch_bounds__(256) void traj_mag(const float* __restrict__ x,
                                                float* __restrict__ mw,
                                                float* __restrict__ inv,
                                                int nwaves) {
  int w = blockIdx.x * 8 + (threadIdx.x >> 5);
  if (w >= nwaves) return;
  int lane = threadIdx.x & 31;
  int b = w / (Sc - 1);
  int j = w % (Sc - 1);
  const float* p0 = x + ((size_t)(b * Sc + j)) * Dc + lane * 32;
  const float* p1 = p0 + Dc;
  float ss = 0.f;
#pragma unroll
  for (int i = 0; i < 8; ++i) {
    float4 a = reinterpret_cast<const float4*>(p0)[i];
    float4 c = reinterpret_cast<const float4*>(p1)[i];
    float dx = c.x - a.x, dy = c.y - a.y, dz = c.z - a.z, dw = c.w - a.w;
    ss += dx * dx + dy * dy + dz * dz + dw * dw;
  }
#pragma unroll
  for (int o = 1; o < 32; o <<= 1) ss += __shfl_xor(ss, o, 32);
  if (lane == 0) {
    float mag = sqrtf(ss);
    mw[w]  = tanhf(mag);
    inv[w] = 1.0f / fmaxf(mag, 1e-8f);
  }
}

// ---------------------------------------------------------------------------
// Trajectory pass 2: windowed recency-weighted sum of normalized diffs.
// ---------------------------------------------------------------------------
__global__ __launch_bounds__(256) void traj_combine(const float* __restrict__ x,
                                                    const float* __restrict__ mw,
                                                    const float* __restrict__ inv,
                                                    float* __restrict__ traj) {
  int s = blockIdx.x, b = blockIdx.y;
  int d = threadIdx.x * 4;
  int ws = s - 8; if (ws < 0) ws = 0;
  int wsz = s - ws; if (wsz < 1) wsz = 1;
  float cw[8]; float denom = 0.f;
  for (int j = ws; j < s; ++j) {
    float rec = (float)(j - ws + 1) / (float)wsz;
    float c = mw[b * (Sc - 1) + j] * rec;
    cw[j - ws] = c; denom += c;
  }
  denom = fmaxf(denom, 1e-8f);
  float4 acc = {0.f, 0.f, 0.f, 0.f};
  for (int j = ws; j < s; ++j) {
    float wgt = (cw[j - ws] / denom) * inv[b * (Sc - 1) + j];
    const float* p0 = x + ((size_t)(b * Sc + j)) * Dc + d;
    float4 a = *reinterpret_cast<const float4*>(p0);
    float4 c = *reinterpret_cast<const float4*>(p0 + Dc);
    acc.x += wgt * (c.x - a.x); acc.y += wgt * (c.y - a.y);
    acc.z += wgt * (c.z - a.z); acc.w += wgt * (c.w - a.w);
  }
  *reinterpret_cast<float4*>(traj + ((size_t)(b * Sc + s)) * Dc + d) = acc;
}

// ---------------------------------------------------------------------------
// WMMA GEMM: C[M,N] = A[M,Kd] * B[N,Kd]^T, A/B f16 row-major, fp32 accum.
// Block tile 128x64, 8 waves. Double-buffered LDS fed by async global->LDS
// B128 copies: issue tile k+1, WMMA on tile k, s_wait_asynccnt + barrier.
// ---------------------------------------------------------------------------
template <bool OUT_F16>
__global__ __launch_bounds__(256) void wmma_gemm_nt(const half_t* __restrict__ A,
                                                    const half_t* __restrict__ B,
                                                    void* __restrict__ Cout,
                                                    int M, int N, int Kd) {
  __shared__ half_t As[2][128][40];
  __shared__ half_t Bs[2][64][40];
  const int tid  = threadIdx.x;
  const int lane = tid & 31;
  const int wid  = tid >> 5;
  const int m0 = blockIdx.y * 128;
  const int n0 = blockIdx.x * 64;
  const int ar = tid >> 1, acz = (tid & 1) * 16;   // A: 16 halves / thread
  const int br = tid >> 2, bcz = (tid & 3) * 8;    // B: 8 halves / thread
  const half_t* aSrc = A + (size_t)(m0 + ar) * Kd + acz;
  const half_t* bSrc = B + (size_t)(n0 + br) * Kd + bcz;

  // issue async copies of K-tile kk into buffer bf (3 x B128 per thread)
  auto issue_tile = [&](int bf, int kk) {
    const half_t* ap = aSrc + (size_t)kk * 32;
    const half_t* bp = bSrc + (size_t)kk * 32;
    async_copy_b128(&As[bf][ar][acz], ap);
    async_copy_b128(&As[bf][ar][acz + 8], ap + 8);
    async_copy_b128(&Bs[bf][br][bcz], bp);
  };

  issue_tile(0, 0);
  wait_async_zero();
  __syncthreads();

  v8f acc[4] = {};
  const int mrow = wid * 16 + (lane & 15);
  const int koff = (lane >> 4) * 16;
  const int nk = Kd / 32;
  int cur = 0;

  for (int kk = 0; kk < nk; ++kk) {
    if (kk + 1 < nk) issue_tile(cur ^ 1, kk + 1);  // overlap with WMMA below
    v16h afrag = *reinterpret_cast<const v16h*>(&As[cur][mrow][koff]);
#pragma unroll
    for (int t = 0; t < 4; ++t) {
      v16h bfrag = *reinterpret_cast<const v16h*>(&Bs[cur][t * 16 + (lane & 15)][koff]);
      acc[t] = __builtin_amdgcn_wmma_f32_16x16x32_f16(
          false, afrag, false, bfrag, (short)0, acc[t], false, false);
    }
    if (kk + 1 < nk) {
      wait_async_zero();
      __syncthreads();
      cur ^= 1;
    }
  }

#pragma unroll
  for (int t = 0; t < 4; ++t) {
#pragma unroll
    for (int r = 0; r < 8; ++r) {
      int row = m0 + wid * 16 + r + ((lane >> 4) << 3);
      int col = n0 + t * 16 + (lane & 15);
      float v = acc[t][r];
      if (OUT_F16) ((half_t*)Cout)[(size_t)row * N + col] = (half_t)v;
      else         ((float*)Cout)[(size_t)row * N + col] = v;
    }
  }
}

// ---------------------------------------------------------------------------
// Splat Gaussian weights for q and k: exp(-0.5*d2/scale^2)*sigmoid(amp),
// stored f16 padded K=16 -> 32 (zeros). One thread per (b,h,s).
// ---------------------------------------------------------------------------
__global__ __launch_bounds__(256) void splat_weights(const half_t* __restrict__ qkv,
                                                     const float* __restrict__ centers,
                                                     const float* __restrict__ log_scales,
                                                     const float* __restrict__ amps,
                                                     half_t* __restrict__ qw,
                                                     half_t* __restrict__ kw) {
  int idx = blockIdx.x * blockDim.x + threadIdx.x;   // (b*H + h)*S + s
  if (idx >= Bc * Hc * Sc) return;
  int s = idx % Sc, bh = idx / Sc, h = bh % Hc;
  int b = bh / Hc;
  const half_t* qv = qkv + ((size_t)(b * Sc + s)) * 3 * Dc + h * HDc;
  const half_t* kv = qv + Dc;
  half_t* qo = qw + (size_t)idx * KP;
  half_t* ko = kw + (size_t)idx * KP;
#pragma unroll 4
  for (int k = 0; k < Kc; ++k) {
    const float* c = centers + (h * Kc + k) * HDc;
    float dq = 0.f, dk = 0.f;
#pragma unroll 8
    for (int d = 0; d < HDc; ++d) {
      float cd = c[d];
      float eq = (float)qv[d] - cd; dq += eq * eq;
      float ek = (float)kv[d] - cd; dk += ek * ek;
    }
    float sc = __expf(log_scales[h * Kc + k]);
    float inv2 = 0.5f / (sc * sc);
    float amp = 1.f / (1.f + __expf(-amps[h * Kc + k]));
    qo[k] = (half_t)(__expf(-dq * inv2) * amp);
    ko[k] = (half_t)(__expf(-dk * inv2) * amp);
  }
#pragma unroll
  for (int k = Kc; k < KP; ++k) { qo[k] = (half_t)0.f; ko[k] = (half_t)0.f; }
}

// ---------------------------------------------------------------------------
// Fused flash splat-attention with STATIC softmax shift (scores in [0,16]):
// per 32-key step: 2 score WMMAs -> 16 v_exp -> 8 packed ds_store_b32 ->
// A-frag reload -> 4 PV WMMAs. No cross-lane reductions in the hot loop.
// ---------------------------------------------------------------------------
__global__ __launch_bounds__(128) void splat_flash_attn(const half_t* __restrict__ qw,
                                                        const half_t* __restrict__ kw,
                                                        const half_t* __restrict__ qkv,
                                                        half_t* __restrict__ attnout) {
  __shared__ half_t P[4][16][32];
  const int lane = threadIdx.x & 31;
  const int wid  = threadIdx.x >> 5;
  const int h = blockIdx.y, b = blockIdx.z;
  const int q0 = (blockIdx.x * 4 + wid) * 16;
  const half_t* qwb = qw + ((size_t)(b * Hc + h)) * Sc * KP;
  const half_t* kwb = kw + ((size_t)(b * Hc + h)) * Sc * KP;
  const half_t* vb  = qkv + ((size_t)b * Sc) * 3 * Dc + 2 * Dc + h * HDc;
  const int koff = (lane >> 4) * 16;
  const float SMAX = 16.0f;  // score upper bound: K * max(qw*kw) = 16 * 1

  v16h aq = *reinterpret_cast<const v16h*>(qwb + (size_t)(q0 + (lane & 15)) * KP + koff);
  v8f acc[4] = {};
  float lpart[8] = {};  // per-lane partial row sums of exp(s - SMAX)
  const int vkey = 2 * (lane & 15) + (lane >> 4);  // interleaved key order

  for (int j0 = 0; j0 < Sc; j0 += 32) {
    v16h kb0 = *reinterpret_cast<const v16h*>(kwb + (size_t)(j0 + (lane & 15)) * KP + koff);
    v16h kb1 = *reinterpret_cast<const v16h*>(kwb + (size_t)(j0 + 16 + (lane & 15)) * KP + koff);
    v8f z = {};
    v8f s0 = __builtin_amdgcn_wmma_f32_16x16x32_f16(false, aq, false, kb0, (short)0, z, false, false);
    v8f s1 = __builtin_amdgcn_wmma_f32_16x16x32_f16(false, aq, false, kb1, (short)0, z, false, false);

#pragma unroll
    for (int r = 0; r < 8; ++r) {
      float p0 = __expf(s0[r] - SMAX);
      float p1 = __expf(s1[r] - SMAX);
      lpart[r] += p0 + p1;
      int prow = r + ((lane >> 4) << 3);
      union { half_t hh[2]; unsigned int u; } pk;
      pk.hh[0] = (half_t)p0; pk.hh[1] = (half_t)p1;
      *reinterpret_cast<unsigned int*>(&P[wid][prow][(lane & 15) * 2]) = pk.u;
    }
    asm volatile("s_wait_dscnt 0" ::: "memory");  // P stores visible before reload
    v16h ap = *reinterpret_cast<const v16h*>(&P[wid][lane & 15][koff]);
#pragma unroll
    for (int t = 0; t < 4; ++t) {
      v16h vf;
#pragma unroll
      for (int s = 0; s < 16; ++s) {
        int c = koff + s;
        vf[s] = vb[(size_t)(j0 + (c >> 1) + (c & 1) * 16) * 3 * Dc + t * 16 + (lane & 15)];
      }
      acc[t] = __builtin_amdgcn_wmma_f32_16x16x32_f16(
          false, ap, false, vf, (short)0, acc[t], false, false);
    }
  }

  // one-time row-sum reduction across the 16-lane groups
#pragma unroll
  for (int r = 0; r < 8; ++r) {
    float s = lpart[r];
#pragma unroll
    for (int o = 1; o < 16; o <<= 1) s += __shfl_xor(s, o, 32);
    lpart[r] = s;
  }

#pragma unroll
  for (int t = 0; t < 4; ++t) {
#pragma unroll
    for (int r = 0; r < 8; ++r) {
      int row = q0 + r + ((lane >> 4) << 3);
      int col = h * HDc + t * 16 + (lane & 15);
      float v = acc[t][r] / fmaxf(lpart[r], 1e-20f);
      attnout[((size_t)(b * Sc + row)) * Dc + col] = (half_t)v;
    }
  }
}

// ---------------------------------------------------------------------------
// Host launcher
// ---------------------------------------------------------------------------
extern "C" void kernel_launch(void* const* d_in, const int* in_sizes, int n_in,
                              void* d_out, int out_size, void* d_ws, size_t ws_size,
                              hipStream_t stream) {
  const float* x      = (const float*)d_in[0];   // [B,S,D]
  const float* qkv_w  = (const float*)d_in[1];   // [3D,D]
  const float* out_w  = (const float*)d_in[2];   // [D,D]
  const float* cent   = (const float*)d_in[3];   // [H,K,hd]
  const float* lsc    = (const float*)d_in[4];   // [H,K]
  const float* amp    = (const float*)d_in[5];   // [H,K]

  const size_t BSD = (size_t)Bc * Sc * Dc;
  float* out_f32  = (float*)d_out;          // [B,S,D]
  float* traj_out = (float*)d_out + BSD;    // [B,S,D]

  uint8_t* w = (uint8_t*)d_ws;
  half_t* x_h    = (half_t*)w; w += BSD * 2;                       // 8 MB
  half_t* wqkv_h = (half_t*)w; w += (size_t)3 * Dc * Dc * 2;       // 6 MB
  half_t* wout_h = (half_t*)w; w += (size_t)Dc * Dc * 2;           // 2 MB
  half_t* qkv_h  = (half_t*)w; w += BSD * 3 * 2;                   // 24 MB
  half_t* qw_h   = (half_t*)w; w += (size_t)Bc * Hc * Sc * KP * 2; // 4 MB
  half_t* kw_h   = (half_t*)w; w += (size_t)Bc * Hc * Sc * KP * 2; // 4 MB
  half_t* ao_h   = (half_t*)w; w += BSD * 2;                       // 8 MB
  float*  mw     = (float*)w;  w += (size_t)Bc * (Sc - 1) * 4;
  float*  invm   = (float*)w;  w += (size_t)Bc * (Sc - 1) * 4;
  (void)ws_size; (void)n_in; (void)in_sizes; (void)out_size;

  // 1) casts to f16
  {
    int n4 = (int)(BSD / 4);
    cast_f32_to_f16<<<(n4 + 255) / 256, 256, 0, stream>>>(x, x_h, n4);
    n4 = 3 * Dc * Dc / 4;
    cast_f32_to_f16<<<(n4 + 255) / 256, 256, 0, stream>>>(qkv_w, wqkv_h, n4);
    n4 = Dc * Dc / 4;
    cast_f32_to_f16<<<(n4 + 255) / 256, 256, 0, stream>>>(out_w, wout_h, n4);
  }

  // 2) trajectories -> second half of d_out
  {
    int nwaves = Bc * (Sc - 1);
    traj_mag<<<(nwaves + 7) / 8, 256, 0, stream>>>(x, mw, invm, nwaves);
    traj_combine<<<dim3(Sc, Bc), 256, 0, stream>>>(x, mw, invm, traj_out);
  }

  // 3) QKV projection: [4096,1024] x [3072,1024]^T -> f16 qkv
  wmma_gemm_nt<true><<<dim3(3 * Dc / 64, Bc * Sc / 128), 256, 0, stream>>>(
      x_h, wqkv_h, qkv_h, Bc * Sc, 3 * Dc, Dc);

  // 4) splat Gaussian weights (K padded to 32)
  splat_weights<<<(Bc * Hc * Sc + 255) / 256, 256, 0, stream>>>(
      qkv_h, cent, lsc, amp, qw_h, kw_h);

  // 5) fused flash splat-attention -> f16 head outputs [B,S,D]
  splat_flash_attn<<<dim3(Sc / 64, Hc, Bc), 128, 0, stream>>>(qw_h, kw_h, qkv_h, ao_h);

  // 6) output projection: [4096,1024] x [1024,1024]^T -> fp32 d_out
  wmma_gemm_nt<false><<<dim3(Dc / 64, Bc * Sc / 128), 256, 0, stream>>>(
      ao_h, wout_h, out_f32, Bc * Sc, Dc, Dc);
}
